// SelfAttentionBlock_23278722744443
// MI455X (gfx1250) — matrix operand order repaired and checked by reading, not compiled, not run on hardware
//
#include <hip/hip_runtime.h>
#include <hip/hip_bf16.h>

// ---- problem constants (from reference) ----
#define Bt 4
#define Cc 512
#define Ss 1024          // H*W = 32*32
#define GRP 8
#define CG  64           // C / GROUPS
#define HDS 16           // heads
#define HD  32           // head dim
#define M_TOT (Bt*Ss)    // 4096 GEMM rows
#define BSTR 40          // padded LDS row stride (elems) for 32-wide B tiles

typedef __attribute__((ext_vector_type(16))) __bf16 v16bf;
typedef __attribute__((ext_vector_type(8)))  __bf16 v8bf;
typedef __attribute__((ext_vector_type(8)))  float  v8f;

__device__ __forceinline__ __bf16 f2bf(float f) {
    union { float f; unsigned u; } x; x.f = f;
    unsigned r = x.u + 0x7FFFu + ((x.u >> 16) & 1u);   // RNE
    unsigned short h = (unsigned short)(r >> 16);
    return __builtin_bit_cast(__bf16, h);
}

// A-operand (16x32 bf16, M x K) per-lane load, row-major source.
//   slots 0..15 -> K = kh*16 + 0..15
__device__ __forceinline__ v16bf loadA(const __bf16* rowp, int kh) {
    v8bf lo = *(const v8bf*)(rowp + kh * 16);
    v8bf hi = *(const v8bf*)(rowp + kh * 16 + 8);
    v16bf a;
#pragma unroll
    for (int i = 0; i < 8; i++) { a[i] = lo[i]; a[8 + i] = hi[i]; }
    return a;
}

__device__ __forceinline__ v8f wmma_bf16(v16bf a, v16bf b, v8f c) {
    return __builtin_amdgcn_wmma_f32_16x16x32_bf16(false, a, false, b,
                                                   (short)0, c, false, false);
}

// CDNA5 async global->LDS copy (16B per lane), tracked by ASYNCcnt.
__device__ __forceinline__ void async_b128_to_lds(const __bf16* g, unsigned lds_byte) {
    asm volatile("global_load_async_to_lds_b128 %0, %1, off"
                 :: "v"(lds_byte), "v"(g) : "memory");
}
__device__ __forceinline__ void wait_async0() {
    asm volatile("s_wait_asynccnt 0" ::: "memory");
}

__device__ __forceinline__ float hredmax16(float v) {
#pragma unroll
    for (int o = 8; o; o >>= 1) v = fmaxf(v, __shfl_xor(v, o, 32));
    return v;
}
__device__ __forceinline__ float hredsum16(float v) {
#pragma unroll
    for (int o = 8; o; o >>= 1) v += __shfl_xor(v, o, 32);
    return v;
}

// ---------------- Kernel 1: group-norm statistics ----------------
__global__ void gn_stats(const float* __restrict__ x, float* __restrict__ stats) {
    int bg = blockIdx.x;                 // 0..31  (b*8+g)
    const float* p = x + (size_t)bg * CG * Ss;
    float s = 0.f, ss = 0.f;
    for (int i = threadIdx.x; i < CG * Ss; i += 256) {
        float v = p[i]; s += v; ss += v * v;
    }
    __shared__ float sh[512];
    sh[threadIdx.x] = s; sh[256 + threadIdx.x] = ss;
    __syncthreads();
    for (int o = 128; o; o >>= 1) {
        if (threadIdx.x < o) {
            sh[threadIdx.x] += sh[threadIdx.x + o];
            sh[256 + threadIdx.x] += sh[256 + threadIdx.x + o];
        }
        __syncthreads();
    }
    if (threadIdx.x == 0) {
        float inv = 1.f / (float)(CG * Ss);
        float mu = sh[0] * inv;
        float var = sh[256] * inv - mu * mu;
        stats[bg * 2] = mu;
        stats[bg * 2 + 1] = rsqrtf(var + 1e-5f);
    }
}

// ------------- Kernel 2: normalize + SiLU, write h[B,S,C] bf16 -------------
__global__ void norm_silu(const float* __restrict__ x, const float* __restrict__ gamma,
                          const float* __restrict__ beta, const float* __restrict__ stats,
                          __bf16* __restrict__ h) {
    int bc = blockIdx.x;                 // 0..2047 (b*512+c)
    int b = bc >> 9, c = bc & 511;
    int g = c >> 6;
    float mu = stats[(b * GRP + g) * 2];
    float rs = stats[(b * GRP + g) * 2 + 1];
    float ga = gamma[c] * rs;
    float be = beta[c] - mu * ga;
    const float* px = x + (size_t)bc * Ss;
    __bf16* ph = h + (size_t)b * Ss * Cc + c;
    for (int s = threadIdx.x; s < Ss; s += 256) {
        float v = px[s] * ga + be;
        float sl = v / (1.f + __expf(-v));   // SiLU
        ph[(size_t)s * Cc] = f2bf(sl);
    }
}

// ---------------- Kernel 3: weight f32 -> bf16 ----------------
__global__ void cvt_w(const float* __restrict__ wq, const float* __restrict__ wp,
                      __bf16* __restrict__ wqb, __bf16* __restrict__ wpb) {
    int idx = blockIdx.x * 256 + threadIdx.x;
    if (idx < 3 * Cc * Cc) wqb[idx] = f2bf(wq[idx]);
    if (idx < Cc * Cc)     wpb[idx] = f2bf(wp[idx]);
}

// ------- Kernel 4: QKV GEMM (M=4096, N=1536, K=512), bf16 WMMA -------
// Block tile 128x64 (8 waves x 16 rows, shared 64-col B tile).
// B tiles staged global->LDS with async-to-LDS, double buffered; A register
// double-buffered so the k-step pipeline overlaps DMA, VMEM and WMMA.
__global__ void qkv_gemm(const __bf16* __restrict__ h, const __bf16* __restrict__ wqb,
                         const float* __restrict__ bqkv,
                         __bf16* __restrict__ qb, __bf16* __restrict__ kb,
                         __bf16* __restrict__ vtb) {
    __shared__ __bf16 Bs[2][64 * BSTR];
    int bm = blockIdx.x / 24, bn = blockIdx.x % 24;
    int w = threadIdx.x >> 5;
    int lane = threadIdx.x & 31;
    int m0 = bm * 128 + w * 16;
    int n0 = bn * 64;
    int mrow = lane & 15, kh = lane >> 4, kl = kh * 16;

    // async staging assignment: thread t moves row n = t/4, 8 elems at koff
    int tn = threadIdx.x >> 2;
    int koff = (threadIdx.x & 3) * 8;
    const __bf16* gB = wqb + (size_t)(n0 + tn) * Cc + koff;
    unsigned ldsBase = (unsigned)(uintptr_t)(&Bs[0][0]);
    unsigned myLds = (unsigned)(tn * BSTR + koff) * 2u;
    const unsigned bufBytes = 64u * BSTR * 2u;

    v8f acc[4] = {v8f{}, v8f{}, v8f{}, v8f{}};
    const __bf16* pa = h + (size_t)(m0 + mrow) * Cc;

    async_b128_to_lds(gB, ldsBase + myLds);          // k-step 0 -> buffer 0
    v16bf a_cur = loadA(pa, kh);

    for (int it = 0; it < 16; ++it) {
        int p = it & 1;
        wait_async0();
        __syncthreads();                             // buffer p fully resident
        v16bf a_nxt = a_cur;
        if (it < 15) {
            async_b128_to_lds(gB + (it + 1) * 32, ldsBase + (p ^ 1) * bufBytes + myLds);
            a_nxt = loadA(pa + (it + 1) * 32, kh);
        }
#pragma unroll
        for (int q = 0; q < 4; q++) {
            v16bf bm_ = *(const v16bf*)&Bs[p][(q * 16 + mrow) * BSTR + kl];
            acc[q] = wmma_bf16(a_cur, bm_, acc[q]);
        }
        a_cur = a_nxt;
    }

    int b = m0 >> 10, sbase = m0 & 1023;
    const float qscale = 0.17677669529663687f;       // 1/sqrt(32)
#pragma unroll
    for (int q = 0; q < 4; q++) {
        int n = n0 + q * 16 + mrow;
        int which = n >> 9;          // 0=q 1=k 2=v
        int cp = n & 511;
        int head = cp >> 5, d = cp & 31;
        int bh = b * HDS + head;
        float bias = bqkv[n];
#pragma unroll
        for (int r = 0; r < 8; r++) {
            int s = sbase + r + 8 * kh;
            float val = acc[q][r] + bias;
            if (which == 0)      qb[((size_t)bh * Ss + s) * HD + d] = f2bf(val * qscale);
            else if (which == 1) kb[((size_t)bh * Ss + s) * HD + d] = f2bf(val);
            else                 vtb[((size_t)bh * HD + d) * Ss + s] = f2bf(val);
        }
    }
}

// ------- Kernel 5: flash attention, one wave per 16-row Q tile -------
__global__ void attn(const __bf16* __restrict__ qb, const __bf16* __restrict__ kb,
                     const __bf16* __restrict__ vtb, __bf16* __restrict__ ob) {
    __shared__ __bf16 ldsP[16 * 32];     // P tile bounce: C-layout -> A-layout
    int bh = blockIdx.x >> 6;
    int s0 = (blockIdx.x & 63) << 4;
    int lane = threadIdx.x & 31;
    int mrow = lane & 15, kh = lane >> 4, kl = kh * 16;

    const __bf16* qp = qb + ((size_t)bh * Ss + s0 + mrow) * HD;
    v16bf aQ = loadA(qp, kh);            // Q tile 16x32 (pre-scaled)

    v8f acc0 = v8f{}, acc1 = v8f{};
    float rM[8], rL[8];
#pragma unroll
    for (int r = 0; r < 8; r++) { rM[r] = -1e30f; rL[r] = 0.f; }

    for (int t0 = 0; t0 < Ss; t0 += 32) {
        v16bf bK0 = *(const v16bf*)(kb + ((size_t)bh * Ss + t0 + mrow) * HD + kl);
        v16bf bK1 = *(const v16bf*)(kb + ((size_t)bh * Ss + t0 + 16 + mrow) * HD + kl);
        v8f z0 = v8f{}, z1 = v8f{};
        v8f sc0 = wmma_bf16(aQ, bK0, z0);
        v8f sc1 = wmma_bf16(aQ, bK1, z1);

#pragma unroll
        for (int r = 0; r < 8; r++) {
            float x0 = sc0[r], x1 = sc1[r];
            float mj = hredmax16(fmaxf(x0, x1));
            float nM = fmaxf(rM[r], mj);
            float al = __expf(rM[r] - nM);
            float p0 = __expf(x0 - nM);
            float p1 = __expf(x1 - nM);
            float rs = hredsum16(p0 + p1);
            rL[r] = rL[r] * al + rs;
            rM[r] = nM;
            acc0[r] *= al;
            acc1[r] *= al;
            int m = r + 8 * kh;
            ldsP[m * 32 + mrow]      = f2bf(p0);
            ldsP[m * 32 + 16 + mrow] = f2bf(p1);
        }
        __syncthreads();
        v16bf aP = loadA(ldsP + mrow * 32, kh);   // P as A-operand (16x32)
        __syncthreads();
        v16bf bV0 = *(const v16bf*)(vtb + ((size_t)bh * HD + mrow) * Ss + t0 + kl);
        v16bf bV1 = *(const v16bf*)(vtb + ((size_t)bh * HD + 16 + mrow) * Ss + t0 + kl);
        acc0 = wmma_bf16(aP, bV0, acc0);
        acc1 = wmma_bf16(aP, bV1, acc1);
    }

    int b = bh >> 4, head = bh & 15;
    __bf16* op = ob + (size_t)b * Ss * Cc + (size_t)head * HD;
#pragma unroll
    for (int r = 0; r < 8; r++) {
        int s = s0 + r + 8 * kh;
        float inv = 1.f / rL[r];
        op[(size_t)s * Cc + mrow]      = f2bf(acc0[r] * inv);
        op[(size_t)s * Cc + 16 + mrow] = f2bf(acc1[r] * inv);
    }
}

// ------- Kernel 6: proj GEMM (M=4096, N=512, K=512) + bias + residual -------
// Same async-LDS double-buffered structure as qkv_gemm.
__global__ void proj_gemm(const __bf16* __restrict__ ob, const __bf16* __restrict__ wpb,
                          const float* __restrict__ bp, const float* __restrict__ x,
                          float* __restrict__ out) {
    __shared__ __bf16 Bs[2][64 * BSTR];
    int bm = blockIdx.x >> 3, bn = blockIdx.x & 7;
    int w = threadIdx.x >> 5;
    int lane = threadIdx.x & 31;
    int m0 = bm * 128 + w * 16;
    int n0 = bn * 64;
    int mrow = lane & 15, kh = lane >> 4, kl = kh * 16;

    int tn = threadIdx.x >> 2;
    int koff = (threadIdx.x & 3) * 8;
    const __bf16* gB = wpb + (size_t)(n0 + tn) * Cc + koff;
    unsigned ldsBase = (unsigned)(uintptr_t)(&Bs[0][0]);
    unsigned myLds = (unsigned)(tn * BSTR + koff) * 2u;
    const unsigned bufBytes = 64u * BSTR * 2u;

    v8f acc[4] = {v8f{}, v8f{}, v8f{}, v8f{}};
    const __bf16* pa = ob + (size_t)(m0 + mrow) * Cc;

    async_b128_to_lds(gB, ldsBase + myLds);
    v16bf a_cur = loadA(pa, kh);

    for (int it = 0; it < 16; ++it) {
        int p = it & 1;
        wait_async0();
        __syncthreads();
        v16bf a_nxt = a_cur;
        if (it < 15) {
            async_b128_to_lds(gB + (it + 1) * 32, ldsBase + (p ^ 1) * bufBytes + myLds);
            a_nxt = loadA(pa + (it + 1) * 32, kh);
        }
#pragma unroll
        for (int q = 0; q < 4; q++) {
            v16bf bm_ = *(const v16bf*)&Bs[p][(q * 16 + mrow) * BSTR + kl];
            acc[q] = wmma_bf16(a_cur, bm_, acc[q]);
        }
        a_cur = a_nxt;
    }

    int b = m0 >> 10, sbase = m0 & 1023;
#pragma unroll
    for (int q = 0; q < 4; q++) {
        int c = n0 + q * 16 + mrow;
        float bias = bp[c];
#pragma unroll
        for (int r = 0; r < 8; r++) {
            int s = sbase + r + 8 * kh;
            size_t idx = ((size_t)(b * Cc + c)) * Ss + s;
            out[idx] = x[idx] + acc[q][r] + bias;
        }
    }
}

extern "C" void kernel_launch(void* const* d_in, const int* in_sizes, int n_in,
                              void* d_out, int out_size, void* d_ws, size_t ws_size,
                              hipStream_t stream) {
    const float* x      = (const float*)d_in[0];
    const float* gamma  = (const float*)d_in[1];
    const float* beta   = (const float*)d_in[2];
    const float* w_qkv  = (const float*)d_in[3];
    const float* b_qkv  = (const float*)d_in[4];
    const float* w_proj = (const float*)d_in[5];
    const float* b_proj = (const float*)d_in[6];
    float* out = (float*)d_out;

    char* ws = (char*)d_ws;
    size_t off = 0;
    float*  stats = (float*)(ws + off); off += 256;
    __bf16* wqb = (__bf16*)(ws + off);  off += (size_t)3 * Cc * Cc * 2;   // 1.5 MB
    __bf16* wpb = (__bf16*)(ws + off);  off += (size_t)Cc * Cc * 2;       // 0.5 MB
    __bf16* hb  = (__bf16*)(ws + off);  off += (size_t)M_TOT * Cc * 2;    // 4 MB
    __bf16* qb  = (__bf16*)(ws + off);  off += (size_t)Bt * HDS * Ss * HD * 2; // 4 MB
    __bf16* kb  = (__bf16*)(ws + off);  off += (size_t)Bt * HDS * Ss * HD * 2; // 4 MB
    __bf16* vtb = (__bf16*)(ws + off);  off += (size_t)Bt * HDS * HD * Ss * 2; // 4 MB
    __bf16* obf = (__bf16*)(ws + off);  off += (size_t)M_TOT * Cc * 2;    // 4 MB

    gn_stats<<<Bt * GRP, 256, 0, stream>>>(x, stats);
    norm_silu<<<Bt * Cc, 256, 0, stream>>>(x, gamma, beta, stats, hb);
    cvt_w<<<(3 * Cc * Cc + 255) / 256, 256, 0, stream>>>(w_qkv, w_proj, wqb, wpb);
    // QKV: 32 M-blocks (128 rows) x 24 N-blocks (64 cols) = 768 blocks, 8 waves each
    qkv_gemm<<<768, 256, 0, stream>>>(hb, wqb, b_qkv, qb, kb, vtb);
    // attention: one wave per (b,head,s-tile) = 4*16*64 = 4096 blocks
    attn<<<Bt * HDS * (Ss / 16), 32, 0, stream>>>(qb, kb, vtb, obf);
    // proj: 32 M-blocks x 8 N-blocks = 256 blocks
    proj_gemm<<<256, 256, 0, stream>>>(obf, wpb, b_proj, x, out);
}
